// Low_Decoder_72516227826047
// MI455X (gfx1250) — matrix operand order repaired and checked by reading, not compiled, not run on hardware
//
#include <hip/hip_runtime.h>
#include <math.h>
#include <stdint.h>

typedef __attribute__((ext_vector_type(16))) _Float16 v16h;
typedef __attribute__((ext_vector_type(8)))  _Float16 h8;
typedef __attribute__((ext_vector_type(8)))  float    v8f;

#define B_ 256
#define N_ 512
#define E_ 128
#define H_ 128
#define C_CLIPF 10.0f
#define NEG_INFF -100000000.0f
#define SCALEF 0.08838834764831845f   // 1/sqrt(128)

__device__ __forceinline__ uint32_t mix32(uint32_t x) {
  x ^= x >> 16; x *= 0x7FEB352Du;
  x ^= x >> 15; x *= 0x846CA68Bu;
  x ^= x >> 16; return x;
}

// ---------------- precompute: h_rest0[j] = b_v[j] + sum_e low_init_w[e]*W_v[e,j]
__global__ void hrest0_kernel(const float* __restrict__ low_init_w,
                              const float* __restrict__ W_v,
                              const float* __restrict__ b_v,
                              float* __restrict__ hrest0) {
  int j = threadIdx.x;             // 128 threads
  float acc = b_v[j];
  for (int e = 0; e < 2 * E_; ++e) acc += low_init_w[e] * W_v[e * E_ + j];
  hrest0[j] = acc;
}

// ---------------- precompute: h_bar[b,j] = b_h[j] + sum_e mean_n(ctx[b,n,e]) * W_h[e,j]
__global__ void hbar_kernel(const float* __restrict__ ctx,
                            const float* __restrict__ W_h,
                            const float* __restrict__ b_h,
                            float* __restrict__ hbar) {
  int b = blockIdx.x, e = threadIdx.x;  // 128 threads
  __shared__ float mean_s[E_];
  const float* p = ctx + ((size_t)b * N_) * E_ + e;
  float s = 0.f;
  for (int n = 0; n < N_; ++n) s += p[(size_t)n * E_];
  mean_s[e] = s * (1.0f / (float)N_);
  __syncthreads();
  float acc = b_h[e];
  for (int k = 0; k < E_; ++k) acc += mean_s[k] * W_h[k * E_ + e];
  hbar[(size_t)b * E_ + e] = acc;
}

// WMMA A-operand (16x32 f16) from a contiguous f16 row (keys row or replicated vec).
__device__ __forceinline__ v16h loadA16(const _Float16* __restrict__ row,
                                        int kk, int hsel) {
  const h8* p0 = (const h8*)(row + kk * 32 + hsel * 16);
  const h8* p1 = (const h8*)(row + kk * 32 + hsel * 16 + 8);
  h8 lo = *p0, hi = *p1;
  v16h a;
#pragma unroll
  for (int i = 0; i < 8; ++i) { a[i] = lo[i]; a[8 + i] = hi[i]; }
  return a;
}

// WMMA B-operand from a pre-swizzled LDS weight image:
// Wsw[(((kk*2)|hsel)*H_ + c)*16 + r] = W[kk*32 + hsel*16 + r][c]  -> 16 contiguous halves.
__device__ __forceinline__ v16h loadB_sw(const _Float16* __restrict__ Wsw,
                                         int kk, int ctbase, int col, int hsel) {
  const _Float16* p = Wsw + (size_t)(((kk << 1) | hsel) * H_ + ctbase + col) * 16;
  h8 lo = *(const h8*)p;
  h8 hi = *(const h8*)(p + 8);
  v16h bb;
#pragma unroll
  for (int i = 0; i < 8; ++i) { bb[i] = lo[i]; bb[8 + i] = hi[i]; }
  return bb;
}

// WMMA B-operand: vector v replicated across the 16 columns (v contiguous f16).
__device__ __forceinline__ v16h loadB_rep(const _Float16* __restrict__ v,
                                          int kk, int hsel) {
  const _Float16* p = v + kk * 32 + hsel * 16;
  h8 lo = *(const h8*)p;
  h8 hi = *(const h8*)(p + 8);
  v16h bb;
#pragma unroll
  for (int i = 0; i < 8; ++i) { bb[i] = lo[i]; bb[8 + i] = hi[i]; }
  return bb;
}

// ---------------- main sequential decoder: one block per batch row, 8 wave32s.
__global__ __launch_bounds__(256)
void decode_kernel(const float* __restrict__ ctx,
                   const float* __restrict__ nodes,
                   const float* __restrict__ maskg,
                   const float* __restrict__ W_v, const float* __restrict__ b_v,
                   const float* __restrict__ W_q, const float* __restrict__ b_q,
                   const float* __restrict__ W_k, const float* __restrict__ b_k,
                   const float* __restrict__ hbar_ws,
                   const float* __restrict__ hrest0_ws,
                   float* __restrict__ out) {
  // LDS (<=320KB/WGP): keys 128K + wk/wv_sw 64K + wq_sw 32K + ~12K misc
  __shared__ _Float16 keysL[N_ * H_];     // f16 keys, resident across all 512 steps
  __shared__ float    wk[E_ * H_];        // W_k f32 (setup); aliased as swizzled W_v f16
  __shared__ _Float16 wq_sw[E_ * H_];     // W_q, pre-swizzled to WMMA B-operand order
  __shared__ float    scores[N_];
  __shared__ float    maskL[N_];
  __shared__ _Float16 qh[H_];             // q in f16 (contiguous)
  __shared__ _Float16 xqh[E_];            // f16(h_bar + h_rest)
  __shared__ _Float16 xvh[2 * E_];        // f16 concat(init_h, h_sel)
  __shared__ float    hbarL[E_], hrestL[E_], inithL[E_];
  __shared__ float    bqL[H_], bvL[E_];
  __shared__ float    red_g[256]; __shared__ int red_i[256];
  __shared__ float    red_m[256]; __shared__ float red_s[256];

  const int b    = blockIdx.x;
  const int t    = threadIdx.x;
  const int lane = t & 31;
  const int wv   = t >> 5;        // wave id 0..7
  const int col  = lane & 15;
  const int hsel = lane >> 4;

  const float* ctxb = ctx + (size_t)b * N_ * E_;

  // -------- setup: async-stage W_k into LDS (CDNA5 async-to-LDS path) --------
  for (int i = t * 4; i < E_ * H_; i += 256 * 4) {
    uint32_t loff = (uint32_t)(uintptr_t)&wk[i];   // LDS offset = low 32 bits of the pointer
    const float* ga = W_k + i;
    asm volatile("global_load_async_to_lds_b128 %0, %1, off"
                 :: "v"(loff), "v"(ga) : "memory");
  }
  // stage W_q swizzled (f16), mask, biases, h state while the async copy flies
  for (int idx = t; idx < E_ * H_; idx += 256) {
    int K = idx >> 7, c = idx & (H_ - 1);
    int kk = K >> 5, rem = K & 31, hs = rem >> 4, r = rem & 15;
    wq_sw[(size_t)(((kk << 1) | hs) * H_ + c) * 16 + r] = (_Float16)W_q[idx];
  }
  for (int n = t; n < N_; n += 256) maskL[n] = maskg[(size_t)b * N_ + n];
  if (t < E_) {
    hbarL[t]  = hbar_ws[(size_t)b * E_ + t];
    hrestL[t] = hrest0_ws[t];
    inithL[t] = 0.f;
    bqL[t]    = b_q[t];
    bvL[t]    = b_v[t];
  }
  asm volatile("s_wait_asynccnt 0x0" ::: "memory");
  __syncthreads();

  // -------- build keys = f16(ctx @ W_k + b_k) directly into LDS via WMMA --------
  for (int tid = wv; tid < 32 * 8; tid += 8) {
    int rt = tid >> 3, ct = tid & 7;
    int n0 = rt * 16, h0 = ct * 16;
    v8f acc = {};
#pragma unroll
    for (int kk = 0; kk < 4; ++kk) {
      const float* pr = ctxb + (size_t)(n0 + col) * E_ + kk * 32 + hsel * 16;
      v16h a;
#pragma unroll
      for (int i = 0; i < 8; ++i) {
        a[i]     = (_Float16)pr[i];
        a[8 + i] = (_Float16)pr[8 + i];
      }
      v16h bb;
#pragma unroll
      for (int j = 0; j < 8; ++j) {
        int K = kk * 32 + hsel * 16 + 2 * j;
        bb[2 * j]     = (_Float16)wk[K * H_ + h0 + col];
        bb[2 * j + 1] = (_Float16)wk[(K + 1) * H_ + h0 + col];
      }
      acc = __builtin_amdgcn_wmma_f32_16x16x32_f16(false, a, false, bb,
                                                   (short)0, acc, false, false);
    }
    float bkc = b_k[h0 + col];
#pragma unroll
    for (int v = 0; v < 8; ++v) {
      int n = n0 + hsel * 8 + v;                // C row = v + 8*hsel
      keysL[n * H_ + h0 + col] = (_Float16)(acc[v] + bkc);
    }
  }
  __syncthreads();

  // -------- reuse W_k staging (64KB) as swizzled W_v f16 (2E x E = 64KB exact) ----
  _Float16* wv_sw = (_Float16*)wk;
  for (int idx = t; idx < 2 * E_ * H_; idx += 256) {
    int K = idx >> 7, c = idx & (H_ - 1);
    int kk = K >> 5, rem = K & 31, hs = rem >> 4, r = rem & 15;
    wv_sw[(size_t)(((kk << 1) | hs) * H_ + c) * 16 + r] = (_Float16)W_v[idx];
  }
  __syncthreads();

  // thread-0 private trackers
  float logp_acc = 0.f, Racc = 0.f;
  float prevx = 0.f, prevy = 0.f, startx = 0.f, starty = 0.f;

  // -------- 512 sequential decode steps --------
  for (int step = 0; step < N_; ++step) {
    // x = f16(h_bar + h_rest)
    if (t < E_) xqh[t] = (_Float16)(hbarL[t] + hrestL[t]);
    __syncthreads();

    // q = x @ W_q + b_q via WMMA (x replicated across A rows; wave w owns col tile w)
    {
      v8f qa = {};
#pragma unroll
      for (int kk = 0; kk < 4; ++kk) {
        v16h a  = loadA16(xqh, kk, hsel);
        v16h bb = loadB_sw(wq_sw, kk, wv * 16, col, hsel);
        qa = __builtin_amdgcn_wmma_f32_16x16x32_f16(false, a, false, bb,
                                                    (short)0, qa, false, false);
      }
      if (hsel == 0) qh[wv * 16 + col] = (_Float16)(qa[0] + bqL[wv * 16 + col]);
    }
    __syncthreads();

    // scores[n] = clip*tanh((keys[n]·q)/sqrt(H)) — WMMA over LDS-resident keys,
    // q replicated across the 16 B-columns (two b128 LDS loads per k-step)
#pragma unroll
    for (int tile = 0; tile < 4; ++tile) {
      int n0 = wv * 64 + tile * 16;
      v8f acc = {};
#pragma unroll
      for (int kk = 0; kk < 4; ++kk) {
        v16h a  = loadA16(keysL + (size_t)(n0 + col) * H_, kk, hsel);
        v16h bq = loadB_rep(qh, kk, hsel);
        acc = __builtin_amdgcn_wmma_f32_16x16x32_f16(false, a, false, bq,
                                                     (short)0, acc, false, false);
      }
      if (col == 0) {
#pragma unroll
        for (int v = 0; v < 8; ++v) {
          int n = n0 + hsel * 8 + v;
          float s = C_CLIPF * tanhf(acc[v] * SCALEF);
          if (maskL[n] > 0.0f) s = NEG_INFF;
          scores[n] = s;
        }
      }
    }
    __syncthreads();

    // Gumbel-max sampling + max (pass 1)
    float s0 = scores[t], s1 = scores[t + 256];
    uint32_t hb = ((uint32_t)b * 0x9E3779B1u) ^ ((uint32_t)step * 0x85EBCA77u);
    float u0 = (float)(mix32(hb ^ (uint32_t)t) >> 8)         * (1.0f / 16777216.0f);
    float u1 = (float)(mix32(hb ^ (uint32_t)(t + 256)) >> 8) * (1.0f / 16777216.0f);
    u0 = fminf(fmaxf(u0, 1e-7f), 0.9999999f);
    u1 = fminf(fmaxf(u1, 1e-7f), 0.9999999f);
    float g0 = s0 - logf(-logf(u0));
    float g1 = s1 - logf(-logf(u1));
    if (g0 >= g1) { red_g[t] = g0; red_i[t] = t; }
    else          { red_g[t] = g1; red_i[t] = t + 256; }
    red_m[t] = fmaxf(s0, s1);
    for (int off = 128; off > 0; off >>= 1) {
      __syncthreads();
      if (t < off) {
        if (red_g[t + off] > red_g[t]) { red_g[t] = red_g[t + off]; red_i[t] = red_i[t + off]; }
        red_m[t] = fmaxf(red_m[t], red_m[t + off]);
      }
    }
    __syncthreads();
    const float M = red_m[0];
    const int   A = red_i[0];

    // logsumexp (pass 2)
    red_s[t] = expf(s0 - M) + expf(s1 - M);
    for (int off = 128; off > 0; off >>= 1) {
      __syncthreads();
      if (t < off) red_s[t] += red_s[t + off];
    }
    __syncthreads();

    if (t == 0) {
      float Z  = red_s[0];
      float sA = scores[A];
      logp_acc += sA - M - logf(Z);
      maskL[A] = 1.0f;
      out[B_ + (size_t)b * N_ + step] = (float)A;   // idxs
      float nx = nodes[((size_t)b * N_ + A) * 2 + 0];
      float ny = nodes[((size_t)b * N_ + A) * 2 + 1];
      if (step == 0) { startx = nx; starty = ny; }
      else {
        float dx = nx - prevx, dy = ny - prevy;
        Racc += sqrtf(dx * dx + dy * dy);
      }
      prevx = nx; prevy = ny;
    }
    __syncthreads();

    // xv = f16 concat(init_h, h_sel)
    if (t < E_) {
      float hs = ctxb[(size_t)A * E_ + t];
      if (step == 0) inithL[t] = hs;
      xvh[t]      = (_Float16)((step == 0) ? hs : inithL[t]);
      xvh[E_ + t] = (_Float16)hs;
    }
    __syncthreads();

    // h_rest = xv @ W_v + b_v via WMMA (K=256: 8 k-steps; wave w owns col tile w)
    {
      v8f ra = {};
#pragma unroll
      for (int kk = 0; kk < 8; ++kk) {
        v16h a  = loadA16(xvh, kk, hsel);
        v16h bb = loadB_sw(wv_sw, kk, wv * 16, col, hsel);
        ra = __builtin_amdgcn_wmma_f32_16x16x32_f16(false, a, false, bb,
                                                    (short)0, ra, false, false);
      }
      if (hsel == 0) hrestL[wv * 16 + col] = ra[0] + bvL[wv * 16 + col];
    }
    __syncthreads();
  }

  if (t == 0) {
    const size_t oR = (size_t)B_ + (size_t)B_ * N_;
    const size_t oS = oR + B_;
    const size_t oE = oS + 2 * B_;
    out[b]              = logp_acc;
    out[oR + b]         = Racc;
    out[oS + 2 * b]     = startx;
    out[oS + 2 * b + 1] = starty;
    out[oE + 2 * b]     = prevx;   // end node = last selected
    out[oE + 2 * b + 1] = prevy;
  }
}

extern "C" void kernel_launch(void* const* d_in, const int* in_sizes, int n_in,
                              void* d_out, int out_size, void* d_ws, size_t ws_size,
                              hipStream_t stream) {
  (void)in_sizes; (void)n_in; (void)out_size; (void)ws_size;
  const float* ctx        = (const float*)d_in[0];
  const float* nodes      = (const float*)d_in[1];
  const float* maskg      = (const float*)d_in[2];
  const float* low_init_w = (const float*)d_in[3];
  const float* W_h        = (const float*)d_in[4];
  const float* b_h        = (const float*)d_in[5];
  const float* W_v        = (const float*)d_in[6];
  const float* b_v        = (const float*)d_in[7];
  const float* W_q        = (const float*)d_in[8];
  const float* b_q        = (const float*)d_in[9];
  const float* W_k        = (const float*)d_in[10];
  const float* b_k        = (const float*)d_in[11];
  float* out = (float*)d_out;

  float* hbar_ws   = (float*)d_ws;          // B_*E_ floats
  float* hrest0_ws = hbar_ws + B_ * E_;     // E_ floats

  hrest0_kernel<<<1, E_, 0, stream>>>(low_init_w, W_v, b_v, hrest0_ws);
  hbar_kernel<<<B_, E_, 0, stream>>>(ctx, W_h, b_h, hbar_ws);
  decode_kernel<<<B_, 256, 0, stream>>>(ctx, nodes, maskg, W_v, b_v, W_q, b_q,
                                        W_k, b_k, hbar_ws, hrest0_ws, out);
}
